// MultiHeadSDPAttentionBlock_11175504904786
// MI455X (gfx1250) — matrix operand order repaired and checked by reading, not compiled, not run on hardware
//
#include <hip/hip_runtime.h>
#include <hip/hip_bf16.h>
#include <math.h>

// MI455X / gfx1250: wave32, WMMA bf16 16x16x32, f32 accumulate.
typedef __bf16 bf16;
typedef __attribute__((ext_vector_type(16))) __bf16 v16bf;
typedef __attribute__((ext_vector_type(8)))  __bf16 v8bf;
typedef __attribute__((ext_vector_type(8)))  float  v8f;
typedef __attribute__((ext_vector_type(8)))  float  v8float;

#define SEQ    2048
#define DMODEL 1024
#define NH     16
#define HD     64

__device__ __forceinline__ v8f zero_v8f() {
  v8f z = {0.f, 0.f, 0.f, 0.f, 0.f, 0.f, 0.f, 0.f};
  return z;
}

__device__ __forceinline__ v8f wmma_bf16(v16bf a, v16bf b, v8f c) {
  // D = A(16x32 bf16) x B(32x16 bf16) + C(16x16 f32)
  return __builtin_amdgcn_wmma_f32_16x16x32_bf16(false, a, false, b, (short)0, c,
                                                 false, false);
}

// B-fragment (32x16): lane n reads row n of a row-major [*,64] bf16 matrix,
// two contiguous 8-half chunks at d0 + 8*hi and d0 + 16 + 8*hi.
__device__ __forceinline__ v16bf load_b_frag(const bf16* row, int d0, int hi) {
  v8bf lo = *(const v8bf*)(row + d0 + 8 * hi);
  v8bf hh = *(const v8bf*)(row + d0 + 16 + 8 * hi);
  return __builtin_shufflevector(lo, hh, 0, 1, 2, 3, 4, 5, 6, 7,
                                 8, 9, 10, 11, 12, 13, 14, 15);
}

// A-fragment (16x32): lane m reads row m, two contiguous 8-half chunks at
// d0 + 8*hi and d0 + 16 + 8*hi (per ISA 16-bit A layout).
__device__ __forceinline__ v16bf load_a_frag(const bf16* row, int d0, int hi) {
  v8bf lo = *(const v8bf*)(row + d0 + 8 * hi);
  v8bf hh = *(const v8bf*)(row + d0 + 16 + 8 * hi);
  return __builtin_shufflevector(lo, hh, 0, 1, 2, 3, 4, 5, 6, 7,
                                 8, 9, 10, 11, 12, 13, 14, 15);
}

// Convert two 8-float chunks of a row into a bf16 A-fragment.
__device__ __forceinline__ v16bf cvt_a_frag(const float* row, int d0, int hi) {
  v8float f0 = *(const v8float*)(row + d0 + 8 * hi);
  v8float f1 = *(const v8float*)(row + d0 + 16 + 8 * hi);
  v16bf a;
#pragma unroll
  for (int j = 0; j < 8; ++j) {
    a[j] = (bf16)f0[j];
    a[8 + j] = (bf16)f1[j];
  }
  return a;
}

// ---------------------------------------------------------------------------
// Kernel 0: transpose + convert weights: [H, d, e] f32 -> [H, e, d] bf16.
// ---------------------------------------------------------------------------
__global__ void wt_transpose_kernel(const float* WQ, const float* WK, const float* WV,
                                    bf16* WTQ, bf16* WTK, bf16* WTV) {
  size_t idx = (size_t)blockIdx.x * blockDim.x + threadIdx.x;
  const size_t per = (size_t)NH * DMODEL * HD;
  int w = (int)(idx / per);
  size_t rem = idx % per;
  int h = (int)(rem / (DMODEL * HD));
  int r2 = (int)(rem % (DMODEL * HD));
  int d = r2 / HD;
  int e = r2 % HD;
  const float* src = (w == 0) ? WQ : (w == 1) ? WK : WV;
  bf16* dst = (w == 0) ? WTQ : (w == 1) ? WTK : WTV;
  dst[(size_t)h * HD * DMODEL + (size_t)e * DMODEL + d] =
      (bf16)src[(size_t)h * DMODEL * HD + (size_t)d * HD + e];
}

// ---------------------------------------------------------------------------
// Kernel 1: projections. Per wave: 32 s-rows x 64 e-cols, loop d (K=32/step).
// Weight B-frags are shared by both s-tiles (2x intensity vs one tile).
// mat 0 -> Qp[b,h,s,e], mat 1 -> Kp[b,h,s,e], mat 2 -> Vt[b,h,e,s] (transposed).
// ---------------------------------------------------------------------------
__global__ void project_kernel(const float* Xq, const float* Xk, const float* Xv,
                               const bf16* WTQ, const bf16* WTK, const bf16* WTV,
                               bf16* Qp, bf16* Kp, bf16* Vt, int Bb) {
  const int bhs = Bb * NH * (SEQ / 256);
  int bx = blockIdx.x;
  int mat = bx / bhs;
  int rem = bx % bhs;
  int bh = rem / (SEQ / 256);
  int sblk = rem % (SEQ / 256);
  int b = bh / NH, h = bh % NH;
  int lane = threadIdx.x & 31;
  int wave = threadIdx.x >> 5;
  int l16 = lane & 15;
  int hi = lane >> 4;
  int s0 = sblk * 256 + wave * 32;

  const float* X = (mat == 0) ? Xq : (mat == 1) ? Xk : Xv;
  const bf16* WT = (mat == 0) ? WTQ : (mat == 1) ? WTK : WTV;

  const float* arow0 = X + ((size_t)b * SEQ + s0 + l16) * DMODEL;
  const float* arow1 = arow0 + (size_t)16 * DMODEL;

  v8f acc[2][4];
#pragma unroll
  for (int u = 0; u < 2; ++u)
#pragma unroll
    for (int et = 0; et < 4; ++et) acc[u][et] = zero_v8f();

  for (int d0 = 0; d0 < DMODEL; d0 += 32) {
    v16bf a0 = cvt_a_frag(arow0, d0, hi);
    v16bf a1 = cvt_a_frag(arow1, d0, hi);
#pragma unroll
    for (int et = 0; et < 4; ++et) {
      const bf16* brow = WT + ((size_t)h * HD + et * 16 + l16) * DMODEL;
      v16bf bfr = load_b_frag(brow, d0, hi);
      acc[0][et] = wmma_bf16(a0, bfr, acc[0][et]);
      acc[1][et] = wmma_bf16(a1, bfr, acc[1][et]);
    }
  }

#pragma unroll
  for (int u = 0; u < 2; ++u) {
    if (mat == 2) {
#pragma unroll
      for (int et = 0; et < 4; ++et)
#pragma unroll
        for (int r = 0; r < 8; ++r) {
          int s = s0 + u * 16 + r + 8 * hi;
          int e = et * 16 + l16;
          Vt[((size_t)bh * HD + e) * SEQ + s] = (bf16)acc[u][et][r];
        }
    } else {
      bf16* dst = (mat == 0) ? Qp : Kp;
#pragma unroll
      for (int et = 0; et < 4; ++et)
#pragma unroll
        for (int r = 0; r < 8; ++r) {
          int s = s0 + u * 16 + r + 8 * hi;
          int e = et * 16 + l16;
          dst[((size_t)bh * SEQ + s) * HD + e] = (bf16)acc[u][et][r];
        }
    }
  }
}

// ---------------------------------------------------------------------------
// Kernel 2: column softmax stats (softmax over QUERY axis — the quirk).
// Per wave: 16 k-rows, loop q in steps of 32 (two independent WMMA chains).
// st = K_tile x Q_tile (transposed scores): lane = q, regs = k.
// Lane-local online (m,l), one shfl_xor merge at the end. Stores m_k, 1/l_k.
// ---------------------------------------------------------------------------
__global__ void stats_kernel(const bf16* Qp, const bf16* Kp, float* Ms, float* Li,
                             int Bb) {
  (void)Bb;
  int bx = blockIdx.x;
  int bh = bx / (SEQ / 128);
  int kblk = bx % (SEQ / 128);
  int lane = threadIdx.x & 31;
  int wave = threadIdx.x >> 5;
  int l16 = lane & 15;
  int hi = lane >> 4;
  int kbase = kblk * 128 + wave * 16;
  const float rscale = 0.02209708691207961f; // 1/sqrt(2048)

  const bf16* krow = Kp + ((size_t)bh * SEQ + kbase + l16) * HD;
  v16bf aK0 = load_a_frag(krow, 0, hi);
  v16bf aK1 = load_a_frag(krow, 32, hi);

  float mrun[8], lrun[8];
#pragma unroll
  for (int r = 0; r < 8; ++r) {
    mrun[r] = -3.0e38f;
    lrun[r] = 0.f;
  }

  for (int q0 = 0; q0 < SEQ; q0 += 32) {
    const bf16* qrowA = Qp + ((size_t)bh * SEQ + q0 + l16) * HD;
    const bf16* qrowB = qrowA + 16 * HD;
    __builtin_prefetch(qrowA + 32 * HD, 0, 3);
    v16bf b0a = load_b_frag(qrowA, 0, hi);
    v16bf b1a = load_b_frag(qrowA, 32, hi);
    v16bf b0b = load_b_frag(qrowB, 0, hi);
    v16bf b1b = load_b_frag(qrowB, 32, hi);
    v8f stA = zero_v8f(), stB = zero_v8f();
    stA = wmma_bf16(aK0, b0a, stA);
    stA = wmma_bf16(aK1, b1a, stA);
    stB = wmma_bf16(aK0, b0b, stB);
    stB = wmma_bf16(aK1, b1b, stB);
#pragma unroll
    for (int r = 0; r < 8; ++r) {
      float v0 = stA[r] * rscale;
      float v1 = stB[r] * rscale;
      float nm = fmaxf(mrun[r], fmaxf(v0, v1));
      lrun[r] = lrun[r] * __expf(mrun[r] - nm) + __expf(v0 - nm) + __expf(v1 - nm);
      mrun[r] = nm;
    }
  }

  // Merge the 16 q-lanes of each half-wave (masks 1,2,4,8 stay within halves).
#pragma unroll
  for (int m = 1; m < 16; m <<= 1) {
#pragma unroll
    for (int r = 0; r < 8; ++r) {
      float om = __shfl_xor(mrun[r], m, 32);
      float ol = __shfl_xor(lrun[r], m, 32);
      float nm = fmaxf(mrun[r], om);
      lrun[r] = lrun[r] * __expf(mrun[r] - nm) + ol * __expf(om - nm);
      mrun[r] = nm;
    }
  }

  if (l16 == 0) {
#pragma unroll
    for (int r = 0; r < 8; ++r) {
      int k = kbase + r + 8 * hi;
      Ms[(size_t)bh * SEQ + k] = mrun[r];
      Li[(size_t)bh * SEQ + k] = 1.0f / lrun[r];
    }
  }
}

// ---------------------------------------------------------------------------
// Kernel 3: out[q,e] = sum_k z[q,k] * V[k,e]. Per wave: 32 q x 64 e, loop k.
// K A-frags and Vt B-frags are shared across both q-tiles (16 WMMAs per
// 256 B/lane of loads). Transposed score D-frag feeds z A-frag directly.
// ---------------------------------------------------------------------------
__global__ void attn_out_kernel(const bf16* Qp, const bf16* Kp, const bf16* Vt,
                                const float* Ms, const float* Li, float* out,
                                int Bb) {
  (void)Bb;
  __shared__ float sm[SEQ];
  __shared__ float sl[SEQ];
  int bx = blockIdx.x;
  int bh = bx / (SEQ / 256);
  int qblk = bx % (SEQ / 256);
  int b = bh / NH, h = bh % NH;

  for (int i = threadIdx.x; i < SEQ; i += 256) {
    sm[i] = Ms[(size_t)bh * SEQ + i];
    sl[i] = Li[(size_t)bh * SEQ + i];
  }
  __syncthreads();

  int lane = threadIdx.x & 31;
  int wave = threadIdx.x >> 5;
  int l16 = lane & 15;
  int hi = lane >> 4;
  int q0 = qblk * 256 + wave * 32;
  const float rscale = 0.02209708691207961f;

  // Loop-invariant Q B-frags for both q-tiles.
  const bf16* qrow0 = Qp + ((size_t)bh * SEQ + q0 + l16) * HD;
  const bf16* qrow1 = qrow0 + 16 * HD;
  v16bf bq0[2], bq1[2];
  bq0[0] = load_b_frag(qrow0, 0, hi);
  bq1[0] = load_b_frag(qrow0, 32, hi);
  bq0[1] = load_b_frag(qrow1, 0, hi);
  bq1[1] = load_b_frag(qrow1, 32, hi);

  v8f acc[2][4];
#pragma unroll
  for (int u = 0; u < 2; ++u)
#pragma unroll
    for (int et = 0; et < 4; ++et) acc[u][et] = zero_v8f();

  for (int k0 = 0; k0 < SEQ; k0 += 32) {
    const bf16* krow0 = Kp + ((size_t)bh * SEQ + k0 + l16) * HD;
    const bf16* krow1 = krow0 + 16 * HD;
    // Issue all shared fragment loads up front (one clause), prefetch next.
    v16bf a00 = load_a_frag(krow0, 0, hi);
    v16bf a01 = load_a_frag(krow0, 32, hi);
    v16bf a10 = load_a_frag(krow1, 0, hi);
    v16bf a11 = load_a_frag(krow1, 32, hi);
    v16bf bv[4];
#pragma unroll
    for (int et = 0; et < 4; ++et) {
      const bf16* vrow =
          Vt + ((size_t)bh * HD + et * 16 + l16) * SEQ + k0 + hi * 8;
      bv[et] = __builtin_shufflevector(*(const v8bf*)vrow, *(const v8bf*)(vrow + 16),
                                       0, 1, 2, 3, 4, 5, 6, 7,
                                       8, 9, 10, 11, 12, 13, 14, 15);
      __builtin_prefetch(vrow + 32, 0, 3);
    }
    __builtin_prefetch(krow0 + 32 * HD, 0, 3);

#pragma unroll
    for (int u = 0; u < 2; ++u) {
      v8f st0 = zero_v8f(), st1 = zero_v8f();
      st0 = wmma_bf16(a00, bq0[u], st0);
      st0 = wmma_bf16(a01, bq1[u], st0);
      st1 = wmma_bf16(a10, bq0[u], st1);
      st1 = wmma_bf16(a11, bq1[u], st1);

      // z A-fragment: halfs 0..7 <- k = k0 + r + 8*hi (st0), 8..15 <- +16 (st1).
      v16bf za;
#pragma unroll
      for (int r = 0; r < 8; ++r) {
        int kk0 = k0 + r + 8 * hi;
        int kk1 = kk0 + 16;
        float z0 = __expf(st0[r] * rscale - sm[kk0]) * sl[kk0];
        float z1 = __expf(st1[r] * rscale - sm[kk1]) * sl[kk1];
        za[r] = (bf16)z0;
        za[8 + r] = (bf16)z1;
      }
#pragma unroll
      for (int et = 0; et < 4; ++et) acc[u][et] = wmma_bf16(za, bv[et], acc[u][et]);
    }
  }

#pragma unroll
  for (int u = 0; u < 2; ++u)
#pragma unroll
    for (int et = 0; et < 4; ++et)
#pragma unroll
      for (int r = 0; r < 8; ++r) {
        int q = q0 + u * 16 + r + 8 * hi;
        int e = et * 16 + l16;
        out[((size_t)b * SEQ + q) * (NH * HD) + h * HD + e] = acc[u][et][r];
      }
}

// ---------------------------------------------------------------------------
extern "C" void kernel_launch(void* const* d_in, const int* in_sizes, int n_in,
                              void* d_out, int out_size, void* d_ws, size_t ws_size,
                              hipStream_t stream) {
  (void)n_in; (void)out_size; (void)ws_size;
  const float* keys    = (const float*)d_in[0];
  const float* queries = (const float*)d_in[1];
  const float* values  = (const float*)d_in[2];
  const float* WQ      = (const float*)d_in[3];
  const float* WK      = (const float*)d_in[4];
  const float* WV      = (const float*)d_in[5];
  float* out = (float*)d_out;
  int Bb = in_sizes[0] / (SEQ * DMODEL);

  char* ws = (char*)d_ws;
  size_t wtBytes = (size_t)NH * HD * DMODEL * sizeof(bf16);    // 2 MB each
  size_t qpBytes = (size_t)Bb * NH * SEQ * HD * sizeof(bf16);  // 8 MB each (B=2)
  size_t stBytes = (size_t)Bb * NH * SEQ * sizeof(float);      // 256 KB each
  bf16* WTQ = (bf16*)(ws);
  bf16* WTK = (bf16*)(ws + wtBytes);
  bf16* WTV = (bf16*)(ws + 2 * wtBytes);
  bf16* Qp  = (bf16*)(ws + 3 * wtBytes);
  bf16* Kp  = (bf16*)(ws + 3 * wtBytes + qpBytes);
  bf16* Vt  = (bf16*)(ws + 3 * wtBytes + 2 * qpBytes);
  float* Msb = (float*)(ws + 3 * wtBytes + 3 * qpBytes);
  float* Lib = (float*)(ws + 3 * wtBytes + 3 * qpBytes + stBytes);

  int wtGrid = (3 * NH * DMODEL * HD) / 256;  // exact multiple
  wt_transpose_kernel<<<wtGrid, 256, 0, stream>>>(WQ, WK, WV, WTQ, WTK, WTV);

  int projGrid = 3 * Bb * NH * (SEQ / 256);
  project_kernel<<<projGrid, 256, 0, stream>>>(queries, keys, values,
                                               WTQ, WTK, WTV, Qp, Kp, Vt, Bb);

  int statsGrid = Bb * NH * (SEQ / 128);
  stats_kernel<<<statsGrid, 256, 0, stream>>>(Qp, Kp, Msb, Lib, Bb);

  int attnGrid = Bb * NH * (SEQ / 256);
  attn_out_kernel<<<attnGrid, 256, 0, stream>>>(Qp, Kp, Vt, Msb, Lib, out, Bb);
}
